// Attention_28003186770644
// MI455X (gfx1250) — matrix operand order repaired and checked by reading, not compiled, not run on hardware
//
#include <hip/hip_runtime.h>
#include <hip/hip_bf16.h>

// ---------------- types / helpers ----------------
typedef __attribute__((ext_vector_type(16))) __bf16 v16bf;
typedef __attribute__((ext_vector_type(8)))  float  v8f;
typedef unsigned short u16;
typedef unsigned int   u32;

union BF16x16 { v16bf v; u16 u[16]; };

__device__ __forceinline__ u16 f2bf(float f) {
  unsigned int u = __float_as_uint(f);
  u += 0x7FFFu + ((u >> 16) & 1u);          // round-to-nearest-even
  return (u16)(u >> 16);
}

__device__ __forceinline__ v8f zero8() {
  v8f z = {0.f,0.f,0.f,0.f,0.f,0.f,0.f,0.f};
  return z;
}

// CDNA5 16-bit A fragment 16x32: lane<16 holds row m=lane, K=0..7 & 16..23;
// lane>=16 holds K=8..15 & 24..31.
__device__ __forceinline__ v16bf loadA_lds(const u16* base, int ldu, int lane) {
  int m = lane & 15, half = lane >> 4;
  const u16* p = base + m * ldu + half * 8;
  BF16x16 r;
#pragma unroll
  for (int e = 0; e < 8; ++e) { r.u[e] = p[e]; r.u[8 + e] = p[16 + e]; }
  return r.v;
}

// CDNA5 16-bit B fragment 32x16: lane<16 holds col n=lane, K=0..7 & 16..23;
// lane>=16 holds K=8..15 & 24..31. Requires B stored K-contiguous per column.
__device__ __forceinline__ v16bf loadB_lds(const u16* base, int ldu, int lane) {
  int n = lane & 15, half = lane >> 4;
  const u16* p = base + n * ldu + half * 8;
  BF16x16 r;
#pragma unroll
  for (int e = 0; e < 8; ++e) { r.u[e] = p[e]; r.u[8 + e] = p[16 + e]; }
  return r.v;
}

// A fragment straight from global f32 (row-major, K contiguous), convert to bf16
__device__ __forceinline__ v16bf loadA_gf32(const float* base, int ld, int lane) {
  int m = lane & 15, half = lane >> 4;
  const float* p = base + (size_t)m * ld + half * 8;
  BF16x16 r;
#pragma unroll
  for (int e = 0; e < 8; ++e) { r.u[e] = f2bf(p[e]); r.u[8 + e] = f2bf(p[16 + e]); }
  return r.v;
}

#define WMMA_BF16(A_, B_, C_) \
  __builtin_amdgcn_wmma_f32_16x16x32_bf16(false, (A_), false, (B_), (short)0, (C_), false, false)

// ---------------- problem constants ----------------
constexpr int Bc = 2, Qc = 1024, PASTc = 1024, Tc = 2048;
constexpr int HIDc = 4096, NHc = 32, NKVc = 8, HDc = 128;
constexpr float SCALEc = 0.08838834764831845f;  // 128^-0.5

// =====================================================================
// Generic GEMM:  C[M][N] = A[M][K] * W[N][K]^T   (f32 in, bf16 WMMA, f32 out)
// Compile-time N,K -> constant-stride addressing; loads expressed as
// uniform base (SGPR) + invariant 32-bit lane offset + immediate.
// block = 256 thr (8 waves), tile 128x128, K-step 32, double-buffered LDS.
// =====================================================================
template <int N, int K>
__global__ void __launch_bounds__(256)
gemm_bf16wmma(const float* __restrict__ A, const float* __restrict__ W,
              float* __restrict__ C) {
  __shared__ u16 As[2][128][40];
  __shared__ u16 Bs[2][128][40];
  int tid  = threadIdx.x;
  int lane = tid & 31;
  int wid  = tid >> 5;
  int wm   = wid >> 2;          // 0..1 : 64 rows each
  int wn   = wid & 3;           // 0..3 : 32 cols each
  int m0   = blockIdx.y * 128;
  int n0   = blockIdx.x * 128;

  const float* Abase = A + (size_t)m0 * K;   // uniform
  const float* Wbase = W + (size_t)n0 * K;   // uniform

  // per-thread staging coordinates: 1024 float4 per 128x32 tile / 256 thr = 4 each
  int srow = tid >> 3;                       // row; rows advance by +32 per i
  int sc4  = (tid & 7) * 4;
  const u32 loff = (u32)srow * K + sc4;      // loop-invariant lane offset

  v8f acc[4][2];
#pragma unroll
  for (int mi = 0; mi < 4; ++mi)
#pragma unroll
    for (int ni = 0; ni < 2; ++ni) acc[mi][ni] = zero8();

  float4 ra[4], rb[4];
  // preload K-step 0 into registers, then LDS buffer 0
#pragma unroll
  for (int i = 0; i < 4; ++i) {
    ra[i] = *(const float4*)(Abase + loff + (u32)(i * 32 * K));
    rb[i] = *(const float4*)(Wbase + loff + (u32)(i * 32 * K));
  }
#pragma unroll
  for (int i = 0; i < 4; ++i) {
    int r = srow + i * 32;
    As[0][r][sc4 + 0] = f2bf(ra[i].x); As[0][r][sc4 + 1] = f2bf(ra[i].y);
    As[0][r][sc4 + 2] = f2bf(ra[i].z); As[0][r][sc4 + 3] = f2bf(ra[i].w);
    Bs[0][r][sc4 + 0] = f2bf(rb[i].x); Bs[0][r][sc4 + 1] = f2bf(rb[i].y);
    Bs[0][r][sc4 + 2] = f2bf(rb[i].z); Bs[0][r][sc4 + 3] = f2bf(rb[i].w);
  }
  __syncthreads();

  constexpr int NK = K / 32;
  for (int k = 0; k < NK; ++k) {
    int buf = k & 1;
    // issue global loads for next K-step before consuming this one
    if (k + 1 < NK) {
      const float* Ak = Abase + (k + 1) * 32;   // uniform scalar advance
      const float* Wk = Wbase + (k + 1) * 32;
#pragma unroll
      for (int i = 0; i < 4; ++i) {
        ra[i] = *(const float4*)(Ak + loff + (u32)(i * 32 * K));
        rb[i] = *(const float4*)(Wk + loff + (u32)(i * 32 * K));
      }
      if (k + 2 < NK) {   // prefetch K-step k+2 into caches
        __builtin_prefetch(Ak + 32 + loff, 0, 1);
        __builtin_prefetch(Wk + 32 + loff, 0, 1);
      }
    }

    v16bf af[4], bfg[2];
#pragma unroll
    for (int mi = 0; mi < 4; ++mi)
      af[mi] = loadA_lds(&As[buf][wm * 64 + mi * 16][0], 40, lane);
#pragma unroll
    for (int ni = 0; ni < 2; ++ni)
      bfg[ni] = loadB_lds(&Bs[buf][wn * 32 + ni * 16][0], 40, lane);
#pragma unroll
    for (int mi = 0; mi < 4; ++mi)
#pragma unroll
      for (int ni = 0; ni < 2; ++ni)
        acc[mi][ni] = WMMA_BF16(af[mi], bfg[ni], acc[mi][ni]);

    __syncthreads();
    if (k + 1 < NK) {
      int nb = buf ^ 1;
#pragma unroll
      for (int i = 0; i < 4; ++i) {
        int r = srow + i * 32;
        As[nb][r][sc4 + 0] = f2bf(ra[i].x); As[nb][r][sc4 + 1] = f2bf(ra[i].y);
        As[nb][r][sc4 + 2] = f2bf(ra[i].z); As[nb][r][sc4 + 3] = f2bf(ra[i].w);
        Bs[nb][r][sc4 + 0] = f2bf(rb[i].x); Bs[nb][r][sc4 + 1] = f2bf(rb[i].y);
        Bs[nb][r][sc4 + 2] = f2bf(rb[i].z); Bs[nb][r][sc4 + 3] = f2bf(rb[i].w);
      }
      __syncthreads();
    }
  }

  int col = lane & 15, half = lane >> 4;
#pragma unroll
  for (int mi = 0; mi < 4; ++mi)
#pragma unroll
    for (int ni = 0; ni < 2; ++ni) {
      // base offset per fragment; v-steps become immediates of N floats
      u32 coff = (u32)(m0 + wm * 64 + mi * 16 + half * 8) * N
               + (u32)(n0 + wn * 32 + ni * 16 + col);
#pragma unroll
      for (int v = 0; v < 8; ++v)
        C[coff + (u32)(v * N)] = acc[mi][ni][v];
    }
}

// =====================================================================
// RoPE on q_proj in place.  threads = B*Q*NH*64
// =====================================================================
__global__ void rope_q_kernel(float* __restrict__ qp,
                              const float* __restrict__ cosT,
                              const float* __restrict__ sinT) {
  int idx = blockIdx.x * blockDim.x + threadIdx.x;
  int d2 = idx & 63;
  int h  = (idx >> 6) & 31;
  int q  = (idx >> 11) & 1023;
  int b  = idx >> 21;
  int pos = PASTc + q;
  float c = cosT[pos * 64 + d2], s = sinT[pos * 64 + d2];
  size_t off = ((size_t)(b * Qc + q)) * HIDc + h * HDc;
  float x1 = qp[off + d2], x2 = qp[off + 64 + d2];
  qp[off + d2]      = x1 * c - x2 * s;
  qp[off + 64 + d2] = x2 * c + x1 * s;
}

// =====================================================================
// RoPE on k_proj + append k,v into present cache.  threads = B*Q*NKV*64
// =====================================================================
__global__ void kv_append_kernel(const float* __restrict__ kp,
                                 const float* __restrict__ vp,
                                 const float* __restrict__ cosT,
                                 const float* __restrict__ sinT,
                                 float* __restrict__ pk, float* __restrict__ pv) {
  int idx = blockIdx.x * blockDim.x + threadIdx.x;
  int d2 = idx & 63;
  int kv = (idx >> 6) & 7;
  int q  = (idx >> 9) & 1023;
  int b  = idx >> 19;
  int pos = PASTc + q;
  float c = cosT[pos * 64 + d2], s = sinT[pos * 64 + d2];
  size_t src = ((size_t)(b * Qc + q)) * (NKVc * HDc) + kv * HDc;
  size_t dst = (((size_t)(b * NKVc + kv)) * Tc + pos) * HDc;
  float x1 = kp[src + d2], x2 = kp[src + 64 + d2];
  pk[dst + d2]      = x1 * c - x2 * s;
  pk[dst + 64 + d2] = x2 * c + x1 * s;
  pv[dst + d2]      = vp[src + d2];
  pv[dst + 64 + d2] = vp[src + 64 + d2];
}

// =====================================================================
// Copy past_key/past_value into present cache.  threads = 2*B*NKV*PAST*HD
// =====================================================================
__global__ void copy_past_kernel(const float* __restrict__ pkin,
                                 const float* __restrict__ pvin,
                                 float* __restrict__ pk, float* __restrict__ pv) {
  int idx = blockIdx.x * blockDim.x + threadIdx.x;
  int sel = idx >> 21;
  int r   = idx & ((1 << 21) - 1);
  int d  = r & 127;
  int t  = (r >> 7) & 1023;
  int kv = (r >> 17) & 7;
  int b  = (r >> 20) & 1;
  size_t src = (((size_t)(b * NKVc + kv)) * PASTc + t) * HDc + d;
  size_t dst = (((size_t)(b * NKVc + kv)) * Tc + t) * HDc + d;
  if (sel == 0) pk[dst] = pkin[src];
  else          pv[dst] = pvin[src];
}

// =====================================================================
// Flash attention.  grid = (Q/64, NH, B), block = 128 (4 waves).
// Each wave owns 16 query rows; block shares staged K/V tiles (32 kv steps).
// Staging addresses: uniform per-tile base (SGPR) + invariant lane offset
// + immediate (krow = 4*i + wave, c4 = 4*lane).
// =====================================================================
__global__ void __launch_bounds__(128)
flash_attn_kernel(const float* __restrict__ qp, const float* __restrict__ pk,
                  const float* __restrict__ pv, const int* __restrict__ amask,
                  float* __restrict__ ctx) {
  __shared__ u16   Ks[32][136];      // [kv][hd]  bf16 (B-frag layout for S)
  __shared__ u16   Vt[128][40];      // [hd][kv]  bf16 (B-frag layout for P*V)
  __shared__ u16   Pl[4][16 * 40];   // per-wave P tile 16x32 (A-frag source)
  __shared__ float Sl[4][16][34];    // per-wave S tile (f32, row-major)
  __shared__ float Al[4][16];        // per-wave per-row broadcast (alpha / 1/l)

  int tid  = threadIdx.x;
  int lane = tid & 31;
  int wave = tid >> 5;
  int qtile = blockIdx.x, h = blockIdx.y, b = blockIdx.z;
  int kvh = h >> 2;                       // NH/NKV = 4
  int qrow0 = qtile * 64 + wave * 16;
  int col = lane & 15, half = lane >> 4;
  int row = col, seg = half;              // softmax-phase ownership

  // preload Q fragments (16 rows x 128 hd), f32 -> bf16
  const float* qbase = qp + ((size_t)(b * Qc + qrow0)) * HIDc + h * HDc;
  v16bf qa[4];
#pragma unroll
  for (int kk = 0; kk < 4; ++kk) qa[kk] = loadA_gf32(qbase + kk * 32, HIDc, lane);

  v8f acc[8];
#pragma unroll
  for (int j = 0; j < 8; ++j) acc[j] = zero8();
  float mrow = -1e30f, lrow = 0.f;        // stats for row (lane&15)

  const float* ksrc0 = pk + ((size_t)(b * NKVc + kvh)) * Tc * HDc;  // uniform
  const float* vsrc0 = pv + ((size_t)(b * NKVc + kvh)) * Tc * HDc;  // uniform
  u16* Pw = &Pl[wave][0];
  int tlim = PASTc + qrow0 + row;         // causal limit for this lane's row

  // invariant per-lane staging offsets: row (4*i + wave), col 4*lane
  const u32 stg  = (u32)wave * HDc + (u32)lane * 4;               // i-step: +4*HDc
  const u32 pfo  = (u32)(tid >> 2) * HDc + (u32)(tid & 3) * 32;   // prefetch lane off

  int tmax = PASTc + qtile * 64 + 63;     // causal: last kv index touched
  for (int t0 = 0; t0 < Tc; t0 += 32) {
    if (t0 > tmax) break;
    const float* kt = ksrc0 + (size_t)t0 * HDc;   // uniform per-tile base
    const float* vt = vsrc0 + (size_t)t0 * HDc;
    // ---- stage K (kv-major) and V (transposed, hd-major) tiles ----
#pragma unroll
    for (int i = 0; i < 8; ++i) {
      int krow = i * 4 + wave;            // 0..31 kv
      int c4   = lane * 4;                // hd
      float4 k4 = *(const float4*)(kt + stg + (u32)(i * 4 * HDc));
      Ks[krow][c4 + 0] = f2bf(k4.x); Ks[krow][c4 + 1] = f2bf(k4.y);
      Ks[krow][c4 + 2] = f2bf(k4.z); Ks[krow][c4 + 3] = f2bf(k4.w);
      float4 v4 = *(const float4*)(vt + stg + (u32)(i * 4 * HDc));
      Vt[c4 + 0][krow] = f2bf(v4.x); Vt[c4 + 1][krow] = f2bf(v4.y);
      Vt[c4 + 2][krow] = f2bf(v4.z); Vt[c4 + 3][krow] = f2bf(v4.w);
    }
    // prefetch next KV tile into caches (global_prefetch_b8)
    if (t0 + 32 < Tc && t0 + 32 <= tmax) {
      __builtin_prefetch(kt + 32 * HDc + pfo, 0, 1);
      __builtin_prefetch(vt + 32 * HDc + pfo, 0, 1);
    }
    __syncthreads();

    // ---- S = Q * K^T  (16 x 32 kv cols) : 8 WMMAs ----
    v8f s0 = zero8(), s1 = zero8();
#pragma unroll
    for (int kk = 0; kk < 4; ++kk) {
      v16bf b0 = loadB_lds(&Ks[0][kk * 32],  136, lane);
      v16bf b1 = loadB_lds(&Ks[16][kk * 32], 136, lane);
      s0 = WMMA_BF16(qa[kk], b0, s0);
      s1 = WMMA_BF16(qa[kk], b1, s1);
    }

    // ---- spill S tile to per-wave LDS (C layout -> row-major) ----
#pragma unroll
    for (int v = 0; v < 8; ++v) {
      Sl[wave][v + half * 8][col]      = s0[v];
      Sl[wave][v + half * 8][col + 16] = s1[v];
    }

    // ---- row-wise online softmax (lane owns row, 16 cols each) ----
    const float* srow = &Sl[wave][row][seg * 16];
    int tbase = t0 + seg * 16;
    int4 am0 = *(const int4*)(amask + b * Tc + tbase);
    int4 am1 = *(const int4*)(amask + b * Tc + tbase + 4);
    int4 am2 = *(const int4*)(amask + b * Tc + tbase + 8);
    int4 am3 = *(const int4*)(amask + b * Tc + tbase + 12);
    int amv[16] = {am0.x, am0.y, am0.z, am0.w, am1.x, am1.y, am1.z, am1.w,
                   am2.x, am2.y, am2.z, am2.w, am3.x, am3.y, am3.z, am3.w};
    float x[16];
#pragma unroll
    for (int j = 0; j < 16; ++j) {
      bool ok = (amv[j] != 0) && (tbase + j <= tlim);
      x[j] = srow[j] * SCALEc + (ok ? 0.f : -10000.f);
    }
    float mx = x[0];
#pragma unroll
    for (int j = 1; j < 16; ++j) mx = fmaxf(mx, x[j]);
    mx = fmaxf(mx, __shfl_xor(mx, 16, 32));          // merge both segments
    float mnew = fmaxf(mrow, mx);
    float a    = __expf(mrow - mnew);
    float rs   = 0.f;
    u16* prow = Pw + row * 40 + seg * 16;
#pragma unroll
    for (int jj = 0; jj < 8; ++jj) {
      float p0 = __expf(x[2 * jj]     - mnew);
      float p1 = __expf(x[2 * jj + 1] - mnew);
      rs += p0 + p1;
      u32 pk2 = (u32)f2bf(p0) | ((u32)f2bf(p1) << 16);
      *(u32*)(prow + 2 * jj) = pk2;                  // packed bf16 pair
    }
    rs += __shfl_xor(rs, 16, 32);                    // merge both segments
    lrow = lrow * a + rs;
    mrow = mnew;
    if (seg == 0) Al[wave][row] = a;                 // broadcast alpha per row

    // ---- rescale accumulators (alpha for rows v + half*8) ----
    float alv[8];
#pragma unroll
    for (int v = 0; v < 8; ++v) alv[v] = Al[wave][v + half * 8];
#pragma unroll
    for (int j = 0; j < 8; ++j)
#pragma unroll
      for (int v = 0; v < 8; ++v) acc[j][v] *= alv[v];

    // ---- O += P * V : 8 WMMAs ----
    v16bf pa = loadA_lds(Pw, 40, lane);
#pragma unroll
    for (int j = 0; j < 8; ++j) {
      v16bf vb = loadB_lds(&Vt[j * 16][0], 40, lane);
      acc[j] = WMMA_BF16(pa, vb, acc[j]);
    }
    __syncthreads();
  }

  // ---- normalize and write ctx [b, q, h*128 + d] ----
  if (seg == 0) Al[wave][row] = 1.0f / lrow;
  float inv[8];
#pragma unroll
  for (int v = 0; v < 8; ++v) inv[v] = Al[wave][v + half * 8];
  // base offset; v-steps become immediates of HID floats
  u32 coff = (u32)(b * Qc + qrow0 + half * 8) * HIDc + (u32)(h * HDc + col);
#pragma unroll
  for (int j = 0; j < 8; ++j)
#pragma unroll
    for (int v = 0; v < 8; ++v)
      ctx[coff + (u32)(v * HIDc) + (u32)(j * 16)] = acc[j][v] * inv[v];
}

// =====================================================================
// launch
// =====================================================================
extern "C" void kernel_launch(void* const* d_in, const int* in_sizes, int n_in,
                              void* d_out, int out_size, void* d_ws, size_t ws_size,
                              hipStream_t stream) {
  (void)in_sizes; (void)n_in; (void)out_size; (void)ws_size;
  const float* hs     = (const float*)d_in[0];
  const int*   amask  = (const int*)d_in[1];
  const float* cosT   = (const float*)d_in[3];
  const float* sinT   = (const float*)d_in[4];
  const float* past_k = (const float*)d_in[5];
  const float* past_v = (const float*)d_in[6];
  const float* q_w    = (const float*)d_in[7];
  const float* k_w    = (const float*)d_in[8];
  const float* v_w    = (const float*)d_in[9];
  const float* o_w    = (const float*)d_in[10];

  constexpr size_t ATTN_SZ = (size_t)Bc * Qc * HIDc;          // 8,388,608
  constexpr size_t KV_SZ   = (size_t)Bc * NKVc * Tc * HDc;    // 4,194,304
  float* attn = (float*)d_out;
  float* pk   = attn + ATTN_SZ;
  float* pv   = pk + KV_SZ;

  float* ws  = (float*)d_ws;
  float* qp  = ws;                                            // 2048x4096
  float* kp  = qp + (size_t)2048 * 4096;                      // 2048x1024
  float* vp  = kp + (size_t)2048 * 1024;                      // 2048x1024
  float* ctx = vp + (size_t)2048 * 1024;                      // 2048x4096

  const int M = Bc * Qc;  // 2048

  gemm_bf16wmma<4096, 4096><<<dim3(4096 / 128, M / 128), 256, 0, stream>>>(hs, q_w, qp);
  gemm_bf16wmma<1024, 4096><<<dim3(1024 / 128, M / 128), 256, 0, stream>>>(hs, k_w, kp);
  gemm_bf16wmma<1024, 4096><<<dim3(1024 / 128, M / 128), 256, 0, stream>>>(hs, v_w, vp);

  rope_q_kernel<<<(Bc * Qc * NHc * 64) / 256, 256, 0, stream>>>(qp, cosT, sinT);
  kv_append_kernel<<<(Bc * Qc * NKVc * 64) / 256, 256, 0, stream>>>(kp, vp, cosT, sinT, pk, pv);
  copy_past_kernel<<<(2 * Bc * NKVc * PASTc * HDc) / 256, 256, 0, stream>>>(past_k, past_v, pk, pv);

  flash_attn_kernel<<<dim3(Qc / 64, NHc, Bc), 128, 0, stream>>>(qp, pk, pv, amask, ctx);

  gemm_bf16wmma<4096, 4096><<<dim3(4096 / 128, M / 128), 256, 0, stream>>>(ctx, o_w, attn);
}
